// NanoVLMGQAttention_68393059221896
// MI455X (gfx1250) — matrix operand order repaired and checked by reading, not compiled, not run on hardware
//
#include <hip/hip_runtime.h>

// ---------------------------------------------------------------------------
// NanoVLM GQA attention for gfx1250 (MI455X, wave32, WMMA bf16 f32-accum)
// GEMMs + attention: LDS-staged tiles via async global->LDS copies
// (ASYNCcnt), double-buffered, software-pipelined register fragments.
// ---------------------------------------------------------------------------

#define DEV static __device__ __forceinline__

typedef __attribute__((ext_vector_type(16))) __bf16 bf16x16;
typedef __attribute__((ext_vector_type(8)))  float  f32x8;
typedef int i32x4 __attribute__((vector_size(16)));   // clang vector, not HIP int4

constexpr int BB   = 2;
constexpr int TT   = 1024;
constexpr int EMB  = 2048;
constexpr int NH   = 32;
constexpr int NKV  = 8;
constexpr int HD   = 64;
constexpr int GRP  = NH / NKV;     // 4
constexpr int MROW = BB * TT;      // 2048
constexpr int KVD  = NKV * HD;     // 512
constexpr float SCALE  = 0.125f;   // 1/sqrt(64)
constexpr float NEGINF = -1.0e30f;

#if defined(__has_builtin)
#if __has_builtin(__builtin_amdgcn_global_load_async_to_lds_b128)
#define HAVE_ASYNC_COPY 1
#endif
#if __has_builtin(__builtin_amdgcn_s_wait_asynccnt)
#define HAVE_WAIT_ASYNC 1
#endif
#endif

union FragU { uint4 u[2]; bf16x16 v; };

DEV unsigned short f32_to_bf16(float f) {
  unsigned int u = __float_as_uint(f);
  unsigned int r = u + 0x7FFFu + ((u >> 16) & 1u);   // round-to-nearest-even
  if ((u & 0x7FFFFFFFu) > 0x7F800000u) r = u | 0x00400000u; // quiet NaN
  return (unsigned short)(r >> 16);
}

DEV f32x8 f32x8_zero() {
  f32x8 v;
#pragma unroll
  for (int i = 0; i < 8; ++i) v[i] = 0.f;
  return v;
}

// A-fragment (16x32 bf16, M x K): per-lane 16 contiguous K values
// K base..base+15 (base = 0 for lanes 0-15, 16 for 16-31).
DEV bf16x16 load_frag_a(const unsigned short* p) {
  FragU f;
  f.u[0] = *(const uint4*)(p);
  f.u[1] = *(const uint4*)(p + 8);
  return f.v;
}

// B-fragment (32x16 bf16, K x N): per-lane 16 contiguous K values
// (K=0..15 for lanes 0-15, K=16..31 for lanes 16-31), N = lane&15.
DEV bf16x16 load_frag_b(const unsigned short* p) {
  FragU f;
  f.u[0] = *(const uint4*)(p);
  f.u[1] = *(const uint4*)(p + 8);
  return f.v;
}

DEV f32x8 wmma_bf16(bf16x16 a, bf16x16 b, f32x8 c) {
  return __builtin_amdgcn_wmma_f32_16x16x32_bf16(
      /*neg_a=*/false, a, /*neg_b=*/false, b,
      /*c_mod=*/(short)0, c, /*reuse_a=*/false, /*reuse_b=*/false);
}

// 16-byte global -> LDS copy (async if the toolchain exposes it)
DEV void async_copy16(const unsigned short* gsrc, unsigned short* ldst) {
#if defined(HAVE_ASYNC_COPY)
  __builtin_amdgcn_global_load_async_to_lds_b128(
      (i32x4*)gsrc, (i32x4*)ldst, /*offset=*/0, /*cpol=*/0);
#else
  *(uint4*)ldst = *(const uint4*)gsrc;
#endif
}

DEV void wait_async_copies() {
#if defined(HAVE_ASYNC_COPY)
#if defined(HAVE_WAIT_ASYNC)
  __builtin_amdgcn_s_wait_asynccnt(0);
#else
  asm volatile("s_wait_asynccnt 0x0" ::: "memory");
#endif
#endif
}

// ---------------------------------------------------------------------------
// fp32 -> bf16 conversion (grid-stride)
// ---------------------------------------------------------------------------
__global__ void cvt_f32_bf16(const float* __restrict__ s,
                             unsigned short* __restrict__ d, int n) {
  int i = blockIdx.x * blockDim.x + threadIdx.x;
  int stride = gridDim.x * blockDim.x;
  for (; i < n; i += stride) d[i] = f32_to_bf16(s[i]);
}

// ---------------------------------------------------------------------------
// C[M,N] (fp32) = A[M,K] (bf16) * W[N,K]^T (bf16)
// block = 256 thr (8 waves); block tile 128(M) x 64(N); wave tile 16 x 64.
// B tile (64 cols x 32 k = 4KB) staged in LDS via async copies, double
// buffered; A frags software-pipelined in registers; all 4 B frags
// preloaded before the WMMA chain (one DS wait, 4 back-to-back WMMAs).
// ---------------------------------------------------------------------------
__global__ __launch_bounds__(256) void gemm_bf16_nt(
    const unsigned short* __restrict__ A, const unsigned short* __restrict__ W,
    float* __restrict__ C, int M, int N, int K) {
  (void)M;
  __shared__ unsigned short Bt[2][64 * 32];   // 2 x 4KB

  const int lane = threadIdx.x & 31;
  const int wave = threadIdx.x >> 5;
  const int lc   = lane & 15;
  const int n0   = blockIdx.x * 64;
  const int m0   = blockIdx.y * 128 + wave * 16;
  const int aoff = (lane < 16) ? 0 : 16;
  const int boff = (lane < 16) ? 0 : 16;
  const size_t arow = (size_t)(m0 + lc) * K;

  // async-copy lane mapping: 256 thr x 16B == one 64x32 bf16 tile
  const int ccol = threadIdx.x >> 2;        // 0..63  (B column within tile)
  const int cseg = (threadIdx.x & 3) * 8;   // 0,8,16,24 (k sub-slice)
  const unsigned short* wsrc = W + (size_t)(n0 + ccol) * K + cseg;
  const int ldsoff = ccol * 32 + cseg;

  // prologue: stage tile k=0
  async_copy16(wsrc, &Bt[0][ldsoff]);
  wait_async_copies();
  __syncthreads();

  f32x8 acc[4];
#pragma unroll
  for (int j = 0; j < 4; ++j) acc[j] = f32x8_zero();

  bf16x16 a_cur = load_frag_a(A + arow + 0 + aoff);

  for (int k = 0; k < K; k += 32) {
    const int cur = (k >> 5) & 1;
    const int nxt = cur ^ 1;
    const bool more = (k + 32) < K;

    // kick off async copy of the next B tile (overlaps WMMAs below)
    if (more) async_copy16(wsrc + (k + 32), &Bt[nxt][ldsoff]);

    // prefetch next A fragment into registers + hint the A stream
    bf16x16 a_nxt = a_cur;
    if (more) {
      a_nxt = load_frag_a(A + arow + (k + 32) + aoff);
      const int kp = (k + 256 < K) ? (k + 256) : k;
      __builtin_prefetch(A + arow + kp, 0, 2);
    }

    const unsigned short* bb = &Bt[cur][0];
    bf16x16 b0 = load_frag_b(bb + (0 * 16 + lc) * 32 + boff);
    bf16x16 b1 = load_frag_b(bb + (1 * 16 + lc) * 32 + boff);
    bf16x16 b2 = load_frag_b(bb + (2 * 16 + lc) * 32 + boff);
    bf16x16 b3 = load_frag_b(bb + (3 * 16 + lc) * 32 + boff);
    acc[0] = wmma_bf16(a_cur, b0, acc[0]);
    acc[1] = wmma_bf16(a_cur, b1, acc[1]);
    acc[2] = wmma_bf16(a_cur, b2, acc[2]);
    acc[3] = wmma_bf16(a_cur, b3, acc[3]);
    a_cur = a_nxt;

    wait_async_copies();
    __syncthreads();
  }

  const int mb = m0 + ((lane < 16) ? 0 : 8);
#pragma unroll
  for (int j = 0; j < 4; ++j) {
    const int n = n0 + j * 16 + lc;
#pragma unroll
    for (int i = 0; i < 8; ++i)
      C[(size_t)(mb + i) * N + n] = acc[j][i];
  }
}

// ---------------------------------------------------------------------------
// RoPE on Q: Qf[B*T, E] (fp32, head-major cols) -> Qbf[B,H,T,D] (bf16)
// ---------------------------------------------------------------------------
__global__ void rope_q_kernel(const float* __restrict__ Qf,
                              const float* __restrict__ cosb,
                              const float* __restrict__ sinb,
                              unsigned short* __restrict__ Qbf) {
  int idx = blockIdx.x * blockDim.x + threadIdx.x;   // 2^21 total
  int d = idx & 31;
  int h = (idx >> 5) & 31;
  int t = (idx >> 10) & 1023;
  int b = (idx >> 20) & 1;
  const float* cr = cosb + (size_t)(b * TT + t) * HD;
  const float* sr = sinb + (size_t)(b * TT + t) * HD;
  const float* qr = Qf + (size_t)(b * TT + t) * EMB + h * HD;
  float x1 = qr[d], x2 = qr[d + 32];
  unsigned short* qo = Qbf + ((size_t)(b * NH + h) * TT + t) * HD;
  qo[d]      = f32_to_bf16(x1 * cr[d]      - x2 * sr[d]);
  qo[d + 32] = f32_to_bf16(x2 * cr[d + 32] + x1 * sr[d + 32]);
}

// ---------------------------------------------------------------------------
// RoPE on K + repack, V transpose-repack:
//   Kf/Vf[B*T, 512] -> Kbf[B,KVH,T,D] bf16, Vt[B,KVH,D,T] bf16
// ---------------------------------------------------------------------------
__global__ void rope_kv_kernel(const float* __restrict__ Kf,
                               const float* __restrict__ Vf,
                               const float* __restrict__ cosb,
                               const float* __restrict__ sinb,
                               unsigned short* __restrict__ Kbf,
                               unsigned short* __restrict__ Vt) {
  int idx = blockIdx.x * blockDim.x + threadIdx.x;   // 2^19 total
  int d  = idx & 31;
  int kh = (idx >> 5) & 7;
  int t  = (idx >> 8) & 1023;
  int b  = (idx >> 18) & 1;
  const float* cr = cosb + (size_t)(b * TT + t) * HD;
  const float* sr = sinb + (size_t)(b * TT + t) * HD;
  const float* krow = Kf + (size_t)(b * TT + t) * KVD + kh * HD;
  float k1 = krow[d], k2 = krow[d + 32];
  unsigned short* ko = Kbf + ((size_t)(b * NKV + kh) * TT + t) * HD;
  ko[d]      = f32_to_bf16(k1 * cr[d]      - k2 * sr[d]);
  ko[d + 32] = f32_to_bf16(k2 * cr[d + 32] + k1 * sr[d + 32]);
  const float* vrow = Vf + (size_t)(b * TT + t) * KVD + kh * HD;
  unsigned short* vo = Vt + (size_t)(b * NKV + kh) * HD * TT;
  vo[(size_t)d * TT + t]        = f32_to_bf16(vrow[d]);
  vo[(size_t)(d + 32) * TT + t] = f32_to_bf16(vrow[d + 32]);
}

// ---------------------------------------------------------------------------
// Flash attention: grid (T/128, H, B), 256 thr = 8 waves, wave = 16 q-rows.
// K (32x64) and V (64x32) step tiles are shared by all 8 waves: staged in
// LDS via async copies, double buffered (copy of step kv0+32 overlaps the
// WMMAs on step kv0). Uniform kv loop; per-wave causal compute guard so
// every wave reaches the barrier. P re-layout C->A via per-wave LDS tile.
// ---------------------------------------------------------------------------
__global__ __launch_bounds__(256) void attn_fused(
    const unsigned short* __restrict__ Qbf, const unsigned short* __restrict__ Kbf,
    const unsigned short* __restrict__ Vt,  const float* __restrict__ gate,
    const float* __restrict__ amask, const int* __restrict__ isvis,
    unsigned short* __restrict__ Abf) {
  __shared__ unsigned short Ktile[2][32 * 64];   // 2 x 4KB (key-major)
  __shared__ unsigned short Vtile[2][64 * 32];   // 2 x 4KB (d-major)
  __shared__ unsigned short pT[8][16 * 32];      // per-wave P tiles

  const int lane = threadIdx.x & 31;
  const int wave = threadIdx.x >> 5;
  const int lc   = lane & 15;
  const int qt = blockIdx.x, h = blockIdx.y, b = blockIdx.z;
  const int kvh  = h / GRP;
  const int qr0  = qt * 128 + wave * 16;
  const int moff = (lane < 16) ? 0 : 8;
  const int aoff = (lane < 16) ? 0 : 16;
  const int boff = (lane < 16) ? 0 : 16;

  const unsigned short* Qh = Qbf + (size_t)(b * NH  + h)   * TT * HD;
  const unsigned short* Kh = Kbf + (size_t)(b * NKV + kvh) * TT * HD;
  const unsigned short* Vh = Vt  + (size_t)(b * NKV + kvh) * HD * TT;

  // async-copy lane mappings (256 thr x 16B == one 4KB tile each)
  const int kkey = threadIdx.x >> 3;           // 0..31 key row of K tile
  const int kseg = (threadIdx.x & 7) * 8;      // d slice
  const int vdim = threadIdx.x >> 2;           // 0..63 d row of V tile
  const int vseg = (threadIdx.x & 3) * 8;      // key slice

  const int qrow = qr0 + lc;
  bf16x16 a0 = load_frag_a(Qh + (size_t)qrow * HD + 0  + aoff);
  bf16x16 a1 = load_frag_a(Qh + (size_t)qrow * HD + 32 + aoff);

  float mr[8], lr[8], g0[8], g1[8];
  f32x8 o0 = f32x8_zero(), o1 = f32x8_zero(), o2 = f32x8_zero(), o3 = f32x8_zero();
#pragma unroll
  for (int i = 0; i < 8; ++i) {
    mr[i] = -3.0e38f;
    lr[i] = 0.f;
    const int q  = qr0 + moff + i;
    const int vq = (isvis[b * TT + q] != 0) ? 1 : 0;
    g0[i] = gate[h * 4 + 2 * vq + 0];   // key is text
    g1[i] = gate[h * 4 + 2 * vq + 1];   // key is vision
  }

  const int kvmax = (qt + 1) * 128;   // block-uniform causal bound

  // prologue: stage step kv0 = 0
  async_copy16(Kh + (size_t)kkey * HD + kseg, &Ktile[0][kkey * 64 + kseg]);
  async_copy16(Vh + (size_t)vdim * TT + vseg, &Vtile[0][vdim * 32 + vseg]);
  wait_async_copies();
  __syncthreads();

  for (int kv0 = 0; kv0 < kvmax; kv0 += 32) {
    const int cur = (kv0 >> 5) & 1;
    const int nxt = cur ^ 1;

    // kick off async copies of the next K/V step tiles
    if (kv0 + 32 < kvmax) {
      async_copy16(Kh + (size_t)(kv0 + 32 + kkey) * HD + kseg,
                   &Ktile[nxt][kkey * 64 + kseg]);
      async_copy16(Vh + (size_t)vdim * TT + (kv0 + 32) + vseg,
                   &Vtile[nxt][vdim * 32 + vseg]);
    }

    if (kv0 < qr0 + 16) {   // this wave still has unmasked keys
      const unsigned short* Kc = &Ktile[cur][0];
      const unsigned short* Vc = &Vtile[cur][0];
      // preload all fragments (one DS wait, back-to-back WMMAs)
      bf16x16 k00 = load_frag_b(Kc + (0  + lc) * 64 + 0  + boff);
      bf16x16 k01 = load_frag_b(Kc + (0  + lc) * 64 + 32 + boff);
      bf16x16 k10 = load_frag_b(Kc + (16 + lc) * 64 + 0  + boff);
      bf16x16 k11 = load_frag_b(Kc + (16 + lc) * 64 + 32 + boff);
      bf16x16 vf0 = load_frag_b(Vc + ( 0 + lc) * 32 + boff);
      bf16x16 vf1 = load_frag_b(Vc + (16 + lc) * 32 + boff);
      bf16x16 vf2 = load_frag_b(Vc + (32 + lc) * 32 + boff);
      bf16x16 vf3 = load_frag_b(Vc + (48 + lc) * 32 + boff);

      // ---- scores: 16x32 tile, 4 wmma ----
      f32x8 s0 = f32x8_zero(), s1 = f32x8_zero();
      s0 = wmma_bf16(a0, k00, s0);
      s0 = wmma_bf16(a1, k01, s0);
      s1 = wmma_bf16(a0, k10, s1);
      s1 = wmma_bf16(a1, k11, s1);

      const int k0 = kv0 + lc, k1 = kv0 + 16 + lc;
      const float am0 = amask[b * TT + k0], am1 = amask[b * TT + k1];
      const int vk0 = (isvis[b * TT + k0] != 0);
      const int vk1 = (isvis[b * TT + k1] != 0);

      float p0[8], p1[8], cm[8];
#pragma unroll
      for (int i = 0; i < 8; ++i) {
        const int q = qr0 + moff + i;
        float v0 = s0[i] * SCALE + (vk0 ? g1[i] : g0[i]);
        float v1 = s1[i] * SCALE + (vk1 ? g1[i] : g0[i]);
        if (k0 > q || am0 <= 0.f) v0 = NEGINF;
        if (k1 > q || am1 <= 0.f) v1 = NEGINF;
        p0[i] = v0; p1[i] = v1;
        cm[i] = fmaxf(v0, v1);
      }
      // row max across the 16 lanes of each half-wave
#pragma unroll
      for (int i = 0; i < 8; ++i) {
        float m = cm[i];
        m = fmaxf(m, __shfl_xor(m, 1, 16));
        m = fmaxf(m, __shfl_xor(m, 2, 16));
        m = fmaxf(m, __shfl_xor(m, 4, 16));
        m = fmaxf(m, __shfl_xor(m, 8, 16));
        cm[i] = m;
      }
      float alpha[8];
#pragma unroll
      for (int i = 0; i < 8; ++i) {
        const float mn = fmaxf(mr[i], cm[i]);
        alpha[i] = __expf(mr[i] - mn);
        mr[i] = mn;
        p0[i] = __expf(p0[i] - mn);
        p1[i] = __expf(p1[i] - mn);
        float s = p0[i] + p1[i];
        s += __shfl_xor(s, 1, 16);
        s += __shfl_xor(s, 2, 16);
        s += __shfl_xor(s, 4, 16);
        s += __shfl_xor(s, 8, 16);
        lr[i] = lr[i] * alpha[i] + s;
        o0[i] *= alpha[i]; o1[i] *= alpha[i]; o2[i] *= alpha[i]; o3[i] *= alpha[i];
      }
      // ---- P: C-layout -> A-layout via per-wave LDS tile (wave-internal,
      // DS in-order, no barrier needed) ----
      unsigned short* tw = &pT[wave][0];
#pragma unroll
      for (int i = 0; i < 8; ++i) {
        tw[(moff + i) * 32 + lc]      = f32_to_bf16(p0[i]);
        tw[(moff + i) * 32 + lc + 16] = f32_to_bf16(p1[i]);
      }
      bf16x16 pa = load_frag_a(&pT[wave][lc * 32 + aoff]);
      // ---- P @ V: 4 back-to-back wmma ----
      o0 = wmma_bf16(pa, vf0, o0);
      o1 = wmma_bf16(pa, vf1, o1);
      o2 = wmma_bf16(pa, vf2, o2);
      o3 = wmma_bf16(pa, vf3, o3);
    }

    wait_async_copies();
    __syncthreads();
  }

  // finalize: y[b,t, h*64+d] bf16 for the output GEMM
  unsigned short* outp = Abf + (size_t)(b * TT) * EMB + h * HD;
#pragma unroll
  for (int i = 0; i < 8; ++i) {
    const int q = qr0 + moff + i;
    const float inv = 1.0f / lr[i];
    unsigned short* orow = outp + (size_t)q * EMB;
    orow[ 0 + lc] = f32_to_bf16(o0[i] * inv);
    orow[16 + lc] = f32_to_bf16(o1[i] * inv);
    orow[32 + lc] = f32_to_bf16(o2[i] * inv);
    orow[48 + lc] = f32_to_bf16(o3[i] * inv);
  }
}

// ---------------------------------------------------------------------------
// host-side launch
// ---------------------------------------------------------------------------
extern "C" void kernel_launch(void* const* d_in, const int* in_sizes, int n_in,
                              void* d_out, int out_size, void* d_ws, size_t ws_size,
                              hipStream_t stream) {
  (void)in_sizes; (void)n_in; (void)out_size; (void)ws_size;
  const float* x     = (const float*)d_in[0];
  const float* cosb  = (const float*)d_in[1];
  const float* sinb  = (const float*)d_in[2];
  const float* amask = (const float*)d_in[3];
  const int*   isvis = (const int*)d_in[4];
  const float* Wq    = (const float*)d_in[5];
  const float* Wk    = (const float*)d_in[6];
  const float* Wv    = (const float*)d_in[7];
  const float* Wo    = (const float*)d_in[8];
  const float* gate  = (const float*)d_in[9];
  float* out = (float*)d_out;

  char* ws = (char*)d_ws;
  size_t off = 0;
  auto carve = [&](size_t bytes) -> void* {
    void* p = ws + off;
    off += (bytes + 255) & ~(size_t)255;
    return p;
  };
  unsigned short* Xb  = (unsigned short*)carve((size_t)MROW * EMB * 2);
  unsigned short* Wqb = (unsigned short*)carve((size_t)EMB * EMB * 2);
  unsigned short* Wkb = (unsigned short*)carve((size_t)KVD * EMB * 2);
  unsigned short* Wvb = (unsigned short*)carve((size_t)KVD * EMB * 2);
  unsigned short* Wob = (unsigned short*)carve((size_t)EMB * EMB * 2);
  float*          Qf  = (float*)carve((size_t)MROW * EMB * 4);
  float*          Kf  = (float*)carve((size_t)MROW * KVD * 4);
  float*          Vf  = (float*)carve((size_t)MROW * KVD * 4);
  unsigned short* Qbf = (unsigned short*)carve((size_t)MROW * EMB * 2);
  unsigned short* Kbf = (unsigned short*)carve((size_t)MROW * KVD * 2);
  unsigned short* Vtb = (unsigned short*)carve((size_t)MROW * KVD * 2);
  unsigned short* Abf = (unsigned short*)carve((size_t)MROW * EMB * 2);

  // fp32 -> bf16
  cvt_f32_bf16<<<1024, 256, 0, stream>>>(x,  Xb,  MROW * EMB);
  cvt_f32_bf16<<<1024, 256, 0, stream>>>(Wq, Wqb, EMB * EMB);
  cvt_f32_bf16<<<1024, 256, 0, stream>>>(Wk, Wkb, KVD * EMB);
  cvt_f32_bf16<<<1024, 256, 0, stream>>>(Wv, Wvb, KVD * EMB);
  cvt_f32_bf16<<<1024, 256, 0, stream>>>(Wo, Wob, EMB * EMB);

  // projections
  gemm_bf16_nt<<<dim3(EMB / 64, MROW / 128), 256, 0, stream>>>(Xb, Wqb, Qf, MROW, EMB, EMB);
  gemm_bf16_nt<<<dim3(KVD / 64, MROW / 128), 256, 0, stream>>>(Xb, Wkb, Kf, MROW, KVD, EMB);
  gemm_bf16_nt<<<dim3(KVD / 64, MROW / 128), 256, 0, stream>>>(Xb, Wvb, Vf, MROW, KVD, EMB);

  // RoPE + repack
  rope_q_kernel <<<(BB * TT * NH  * 32) / 256, 256, 0, stream>>>(Qf, cosb, sinb, Qbf);
  rope_kv_kernel<<<(BB * TT * NKV * 32) / 256, 256, 0, stream>>>(Kf, Vf, cosb, sinb, Kbf, Vtb);

  // flash attention
  attn_fused<<<dim3(TT / 128, NH, BB), 256, 0, stream>>>(Qbf, Kbf, Vtb, gate, amask, isvis, Abf);

  // output projection -> fp32 out
  gemm_bf16_nt<<<dim3(EMB / 64, MROW / 128), 256, 0, stream>>>(Abf, Wob, out, MROW, EMB, EMB);
}
